// RoutingRNNCellBase_44126493999766
// MI455X (gfx1250) — matrix operand order repaired and checked by reading, not compiled, not run on hardware
//
#include <hip/hip_runtime.h>
#include <hip/hip_bf16.h>

// ---------------------------------------------------------------------------
// RoutingRNNCell: dense i2h GEMM + 2-stage routed expert GEMM (MoE-style).
// bf16 WMMA (v_wmma_f32_16x16x32_bf16), fp32 accumulate.
// Block tile 64x128xK32, 8 waves, 32x32 C-tile per wave (4 wmma / K-step).
// Double-buffered LDS, K-loop unrolled x2 (static buffer indices).
// ---------------------------------------------------------------------------

typedef __attribute__((ext_vector_type(16))) __bf16 v16bf;
typedef __attribute__((ext_vector_type(8)))  float  v8f;

#define NB      4096      // batch
#define PROJ    4096
#define OUTW    8192
#define BM      64
#define BN      128
#define BK      32
#define LDSK    40        // BK + 8 pad; row stride 80B (16B-aligned)
#define MAXT    72        // worst-case row tiles: 64 + 7 experts + slack

// workspace meta layout (int offsets, after 32MB bf16 intermediate)
#define RM0     0                  // rowMap stage0 [4096]
#define RM1     4096               // rowMap stage1 [4096]
#define TE0     8192               // tileExpert0 [MAXT]
#define TB0     (TE0 + MAXT)
#define TEND0   (TB0 + MAXT)
#define TE1     (TEND0 + MAXT)
#define TB1     (TE1 + MAXT)
#define TEND1   (TB1 + MAXT)
#define NT0     (TEND1 + MAXT)
#define NT1     (NT0 + 1)

union PK2 { __bf16 h[2]; unsigned u; };

// ---------------------------------------------------------------------------
__global__ __launch_bounds__(256)
void bucket_kernel(const int* __restrict__ actions, int* __restrict__ meta) {
    __shared__ int cnt0[8], cnt1[8], off0[8], off1[8], cur0[8], cur1[8];
    int tid = threadIdx.x;
    if (tid < 8) { cnt0[tid] = 0; cnt1[tid] = 0; }
    __syncthreads();
    for (int r = tid; r < NB; r += 256) {
        int a0 = actions[r], a1 = actions[NB + r];
        if (a0 && a1) { atomicAdd(&cnt0[a0], 1); atomicAdd(&cnt1[a1], 1); }
    }
    __syncthreads();
    if (tid == 0) {
        int o = 0;
        for (int e = 0; e < 8; ++e) { off0[e] = o; o += cnt0[e]; }
        o = 0;
        for (int e = 0; e < 8; ++e) { off1[e] = o; o += cnt1[e]; }
        int idx = 0;
        for (int e = 1; e < 8; ++e) {
            int c = cnt0[e], base = off0[e];
            for (int t = 0; t < c; t += BM) {
                meta[TE0 + idx] = e; meta[TB0 + idx] = base + t;
                meta[TEND0 + idx] = base + c; ++idx;
            }
        }
        meta[NT0] = idx;
        idx = 0;
        for (int e = 1; e < 8; ++e) {
            int c = cnt1[e], base = off1[e];
            for (int t = 0; t < c; t += BM) {
                meta[TE1 + idx] = e; meta[TB1 + idx] = base + t;
                meta[TEND1 + idx] = base + c; ++idx;
            }
        }
        meta[NT1] = idx;
    }
    __syncthreads();
    if (tid < 8) { cur0[tid] = off0[tid]; cur1[tid] = off1[tid]; }
    __syncthreads();
    for (int r = tid; r < NB; r += 256) {
        int a0 = actions[r], a1 = actions[NB + r];
        if (a0 && a1) {
            int p0 = atomicAdd(&cur0[a0], 1); meta[RM0 + p0] = r;
            int p1 = atomicAdd(&cur1[a1], 1); meta[RM1 + p1] = r;
        }
    }
}

// Zero the h2h half of the output for masked rows (every call).
__global__ __launch_bounds__(256)
void zero_inactive_kernel(const int* __restrict__ actions, float* __restrict__ out) {
    int row = blockIdx.x;
    int a0 = actions[row], a1 = actions[NB + row];
    if (a0 != 0 && a1 != 0) return;
    float4* o = (float4*)(out + (long)row * OUTW + PROJ);
    float4 z = make_float4(0.f, 0.f, 0.f, 0.f);
    for (int i = threadIdx.x; i < PROJ / 4; i += 256) o[i] = z;
}

// ---------------------------------------------------------------------------
// Tiled bf16-WMMA GEMM with optional row gather/scatter (grouped-expert mode).
//   C[row, colOff+col] = act( A[row,:] * W_e[:,col] + bias_e[col] )
// 8 waves as 2x4; each wave computes a 32x32 C tile (2x2 fragments, 4 wmma).
// ---------------------------------------------------------------------------
template <bool A_BF16, bool RELU, bool OUT_BF16>
__global__ __launch_bounds__(256)
void gemm_routed(const void* __restrict__ Aptr, int Kdim,
                 const float* __restrict__ Wbase, long wExpertStride, int ldW,
                 const float* __restrict__ biasBase, int biasExpertStride,
                 void* __restrict__ Outp, int ldOut, int outColOff,
                 const int* __restrict__ rowMap,
                 const int* __restrict__ tileExpert,
                 const int* __restrict__ tileBase,
                 const int* __restrict__ tileEnd,
                 const int* __restrict__ numTiles)
{
    const int ty = blockIdx.y;
    if (numTiles && ty >= *numTiles) return;

    int expert, rBase, rEnd;
    if (tileExpert) { expert = tileExpert[ty]; rBase = tileBase[ty]; rEnd = tileEnd[ty]; }
    else            { expert = 0;              rBase = ty * BM;      rEnd = NB; }

    const float* W    = Wbase    + (long)expert * wExpertStride;
    const float* bias = biasBase + (long)expert * biasExpertStride;
    const int colTile = blockIdx.x * BN;

    __shared__ __bf16 As[2][BM][LDSK];
    __shared__ __bf16 Bs[2][BN][LDSK];

    const int tid  = threadIdx.x;
    const int wid  = tid >> 5;
    const int lane = tid & 31;
    const int wr   = wid & 1;        // wave's 32-row strip within BM
    const int wc   = wid >> 1;       // wave's 32-col strip within BN
    const int lh   = lane >> 4;      // lane half (ISA fragment layout)
    const int ln   = lane & 15;

    // A staging: 4 threads/row, 8 consecutive K each (16B)
    const int loadRow = tid >> 2;
    const int loadK0  = (tid & 3) * 8;
    const int gRowIdx = rBase + loadRow;
    const bool rowValid = gRowIdx < rEnd;
    const int aRow = rowValid ? (rowMap ? rowMap[gRowIdx] : gRowIdx) : 0;

    // B staging: each thread: 2 consecutive K rows x 8 consecutive N.
    // Commit: 8x (v_cvt_pk_bf16 -> ds_store_b32), K-pair contiguous in Bs.
    const int bK  = (tid >> 4) * 2;     // 0,2,..,30
    const int bN0 = (tid & 15) * 8;     // 0..120

    const float*  Afp = (const float*)Aptr;
    const __bf16* Abf = (const __bf16*)Aptr;

    // register staging between global load and LDS commit
    uint4  ra;                              // A strip (packed bf16)
    float4 rbk0a, rbk0b, rbk1a, rbk1b;      // B: row k (8 fl) + row k+1 (8 fl)

    auto GLOAD = [&](int kt) {
        if (A_BF16) {
            uint4 u = make_uint4(0u, 0u, 0u, 0u);
            if (rowValid)
                u = *(const uint4*)(Abf + (long)aRow * Kdim + kt + loadK0);
            ra = u;
        } else {
            uint4 u = make_uint4(0u, 0u, 0u, 0u);
            if (rowValid) {
                const float4* s = (const float4*)(Afp + (long)aRow * Kdim + kt + loadK0);
                float4 fa0 = s[0], fa1 = s[1];
                // fresh PK2 per pair, inline -> v_cvt_pk_bf16_f32 vD, vA, vB
                { PK2 p; p.h[0] = (__bf16)fa0.x; p.h[1] = (__bf16)fa0.y; u.x = p.u; }
                { PK2 p; p.h[0] = (__bf16)fa0.z; p.h[1] = (__bf16)fa0.w; u.y = p.u; }
                { PK2 p; p.h[0] = (__bf16)fa1.x; p.h[1] = (__bf16)fa1.y; u.z = p.u; }
                { PK2 p; p.h[0] = (__bf16)fa1.z; p.h[1] = (__bf16)fa1.w; u.w = p.u; }
            }
            ra = u;
        }
        const float* r0 = W + (long)(kt + bK) * ldW + colTile + bN0;
        rbk0a = ((const float4*)r0)[0]; rbk0b = ((const float4*)r0)[1];
        const float* r1 = r0 + ldW;
        rbk1a = ((const float4*)r1)[0]; rbk1b = ((const float4*)r1)[1];
    };

    auto LCOMMIT = [&](int b) {
        *(uint4*)(&As[b][loadRow][loadK0]) = ra;      // single ds_store_b128
        float k0[8] = { rbk0a.x, rbk0a.y, rbk0a.z, rbk0a.w,
                        rbk0b.x, rbk0b.y, rbk0b.z, rbk0b.w };
        float k1[8] = { rbk1a.x, rbk1a.y, rbk1a.z, rbk1a.w,
                        rbk1b.x, rbk1b.y, rbk1b.z, rbk1b.w };
#pragma unroll
        for (int j = 0; j < 8; ++j) {
            PK2 p; p.h[0] = (__bf16)k0[j]; p.h[1] = (__bf16)k1[j];
            *(unsigned*)(&Bs[b][bN0 + j][bK]) = p.u;  // ds_store_b32 (pairs fuse)
        }
    };

    v8f acc00 = {}, acc01 = {}, acc10 = {}, acc11 = {};

    auto COMPUTE = [&](int b) {
        // A fragments: rows m = 32*wr + 16*rs + ln
        v16bf a0, a1;
        const int m0 = wr * 32 + ln;
#pragma unroll
        for (int j = 0; j < 8; ++j) a0[j]     = As[b][m0][8 * lh + j];
#pragma unroll
        for (int j = 0; j < 8; ++j) a0[8 + j] = As[b][m0][16 + 8 * lh + j];
        const int m1 = m0 + 16;
#pragma unroll
        for (int j = 0; j < 8; ++j) a1[j]     = As[b][m1][8 * lh + j];
#pragma unroll
        for (int j = 0; j < 8; ++j) a1[8 + j] = As[b][m1][16 + 8 * lh + j];
        // B fragments: cols n = 32*wc + 16*cs + ln; same K order as A
        v16bf b0, b1;
        const int n0 = wc * 32 + ln;
#pragma unroll
        for (int j = 0; j < 8; ++j) b0[j]     = Bs[b][n0][8 * lh + j];
#pragma unroll
        for (int j = 0; j < 8; ++j) b0[8 + j] = Bs[b][n0][16 + 8 * lh + j];
#pragma unroll
        for (int j = 0; j < 8; ++j) b1[j]     = Bs[b][n0 + 16][8 * lh + j];
#pragma unroll
        for (int j = 0; j < 8; ++j) b1[8 + j] = Bs[b][n0 + 16][16 + 8 * lh + j];

        acc00 = __builtin_amdgcn_wmma_f32_16x16x32_bf16(false, a0, false, b0,
                                                        (short)0, acc00, false, false);
        acc01 = __builtin_amdgcn_wmma_f32_16x16x32_bf16(false, a0, false, b1,
                                                        (short)0, acc01, false, false);
        acc10 = __builtin_amdgcn_wmma_f32_16x16x32_bf16(false, a1, false, b0,
                                                        (short)0, acc10, false, false);
        acc11 = __builtin_amdgcn_wmma_f32_16x16x32_bf16(false, a1, false, b1,
                                                        (short)0, acc11, false, false);
    };

    // K-loop, unrolled x2 over the double buffer (Kdim is a multiple of 2*BK).
    GLOAD(0);
    LCOMMIT(0);
    for (int kt = 0; kt < Kdim; kt += 2 * BK) {
        __syncthreads();
        GLOAD(kt + BK);                    // overlap with COMPUTE(0)
        COMPUTE(0);
        LCOMMIT(1);
        __syncthreads();
        const bool more = (kt + 2 * BK) < Kdim;
        if (more) GLOAD(kt + 2 * BK);      // overlap with COMPUTE(1)
        COMPUTE(1);
        if (more) LCOMMIT(0);
    }

    // Epilogue. C layout: lane(n=ln, half=lh), vgpr v -> row 8*lh + v (+16*rs)
    const int col0 = colTile + wc * 32 + ln;
    const float bias0 = bias[col0];
    const float bias1 = bias[col0 + 16];
#pragma unroll
    for (int rs = 0; rs < 2; ++rs) {
        const v8f& c0 = rs ? acc10 : acc00;
        const v8f& c1 = rs ? acc11 : acc01;
#pragma unroll
        for (int v = 0; v < 8; ++v) {
            const int localRow = wr * 32 + rs * 16 + lh * 8 + v;
            const int gi = rBase + localRow;
            if (gi >= rEnd) continue;
            const int row = rowMap ? rowMap[gi] : gi;
            float v0 = c0[v] + bias0;
            float v1 = c1[v] + bias1;
            if (RELU) { v0 = fmaxf(v0, 0.0f); v1 = fmaxf(v1, 0.0f); }
            if (OUT_BF16) {
                __bf16* o = (__bf16*)Outp + (long)row * ldOut + outColOff;
                o[col0]      = (__bf16)v0;
                o[col0 + 16] = (__bf16)v1;
            } else {
                float* o = (float*)Outp + (long)row * ldOut + outColOff;
                o[col0]      = v0;
                o[col0 + 16] = v1;
            }
        }
    }
}

// ---------------------------------------------------------------------------
extern "C" void kernel_launch(void* const* d_in, const int* in_sizes, int n_in,
                              void* d_out, int out_size, void* d_ws, size_t ws_size,
                              hipStream_t stream) {
    const float* x      = (const float*)d_in[0];   // [4096,1024]
    const float* h      = (const float*)d_in[1];   // [4096,1024]
    const int*   act    = (const int*)  d_in[2];   // [2,4096]
    const float* W_i2h  = (const float*)d_in[3];   // [1024,4096]
    const float* b_i2h  = (const float*)d_in[4];   // [4096]
    const float* W_exp0 = (const float*)d_in[5];   // [8,1024,4096]
    const float* b_exp0 = (const float*)d_in[6];   // [8,4096]
    const float* W_exp1 = (const float*)d_in[7];   // [8,4096,4096]
    const float* b_exp1 = (const float*)d_in[8];   // [8,4096]
    float* out = (float*)d_out;                    // [4096,8192]

    __bf16* inter = (__bf16*)d_ws;                               // 32 MB bf16 [4096,4096]
    int* meta = (int*)((char*)d_ws + (size_t)NB * PROJ * 2);     // bucket tables

    (void)in_sizes; (void)n_in; (void)out_size; (void)ws_size;

    // 1) bucket active rows by expert (both depths)
    bucket_kernel<<<1, 256, 0, stream>>>(act, meta);

    // 2) dense i2h: out[:, 0:4096] = x @ W_i2h + b_i2h
    gemm_routed<false, false, false><<<dim3(PROJ / BN, NB / BM), 256, 0, stream>>>(
        x, 1024, W_i2h, 0, PROJ, b_i2h, 0,
        out, OUTW, 0, nullptr, nullptr, nullptr, nullptr, nullptr);

    // 3) zero h2h half for masked rows
    zero_inactive_kernel<<<NB, 256, 0, stream>>>(act, out);

    // 4) depth-0 grouped experts: inter[row] = relu(h[row] @ W_exp0[a0] + b_exp0[a0])
    gemm_routed<false, true, true><<<dim3(PROJ / BN, MAXT), 256, 0, stream>>>(
        h, 1024, W_exp0, (long)1024 * PROJ, PROJ, b_exp0, PROJ,
        inter, PROJ, 0,
        meta + RM0, meta + TE0, meta + TB0, meta + TEND0, meta + NT0);

    // 5) depth-1 grouped experts: out[row, 4096:] = inter[row] @ W_exp1[a1] + b_exp1[a1]
    gemm_routed<true, false, false><<<dim3(PROJ / BN, MAXT), 256, 0, stream>>>(
        inter, PROJ, W_exp1, (long)PROJ * PROJ, PROJ, b_exp1, PROJ,
        out, OUTW, PROJ,
        meta + RM1, meta + TE1, meta + TB1, meta + TEND1, meta + NT1);
}
